// IterativeMoE_7670811590875
// MI455X (gfx1250) — compile-verified, matrix operand order checked by reading, not run on hardware
//
#include <hip/hip_runtime.h>

// ---------------- problem constants ----------------
#define BTOK   2048
#define NEXP   16
#define KTOP   4
#define DIN    1536
#define HID    1024
#define D_SF   512
#define D_OF   512
#define D_PPF  256
#define D_PVF  256
#define N_OBJ  36
#define N_PRED 132
#define C_TOT  204          // 36 + 36 + 132
#define C_PAD  256          // padded head count (multiple of 128 for GEMM tiles)
#define EPS64  2.220446049250313e-16f

typedef __bf16  bf16x8  __attribute__((ext_vector_type(8)));
typedef __bf16  bf16x16 __attribute__((ext_vector_type(16)));
typedef float   f32x4   __attribute__((ext_vector_type(4)));
typedef float   f32x8   __attribute__((ext_vector_type(8)));
typedef int     v4i     __attribute__((vector_size(16)));

// ---------------- CDNA5 async global->LDS path (compile-probed) ----------------
#if defined(__gfx1250__) && __has_builtin(__builtin_amdgcn_global_load_async_to_lds_b128)
#define HAVE_ASYNC 1
#define CP16(g, l) __builtin_amdgcn_global_load_async_to_lds_b128( \
        (v4i*)(g), (v4i*)(l), 0, 0)
#if __has_builtin(__builtin_amdgcn_s_wait_asynccnt)
#define S_WAIT_ASYNC(n) __builtin_amdgcn_s_wait_asynccnt(n)
#else
#define S_WAIT_ASYNC(n) asm volatile("s_wait_asynccnt %0" :: "n"(n) : "memory")
#endif
#else
#define HAVE_ASYNC 0
#define CP16(g, l) (*(bf16x8*)(l) = *(const bf16x8*)(g))
#define S_WAIT_ASYNC(n) do {} while (0)
#endif

// ---------------- workspace layout (bytes) ----------------
#define OFF_XB     0LL                                   // [B, D] bf16
#define OFF_GATES  (OFF_XB    + (long long)BTOK*DIN*2)   // [B, E] f32
#define OFF_W1T    (OFF_GATES + (long long)BTOK*NEXP*4)  // [E, H, D] bf16
#define OFF_W2T    (OFF_W1T   + (long long)NEXP*HID*DIN*2)   // [E, H, H] bf16
#define OFF_WHT    (OFF_W2T   + (long long)NEXP*HID*HID*2)   // [E, C_PAD, H] bf16
#define OFF_BH     (OFF_WHT   + (long long)NEXP*C_PAD*HID*2) // [E, C_PAD] f32
#define OFF_H1     (OFF_BH    + (long long)NEXP*C_PAD*4)     // [E, B, H] bf16
#define OFF_H2     (OFF_H1    + (long long)NEXP*BTOK*HID*2)  // [E, B, H] bf16
#define OFF_HL     (OFF_H2    + (long long)NEXP*BTOK*HID*2)  // [E, B, C_PAD] f32

// =====================================================================
// Kernel 1: concat -> bf16 x, noisy top-k gating
// =====================================================================
__global__ __launch_bounds__(128)
void gate_kernel(const float* __restrict__ sf,  const float* __restrict__ of,
                 const float* __restrict__ ppf, const float* __restrict__ pvf,
                 const float* __restrict__ w_gate, const float* __restrict__ w_noise,
                 const float* __restrict__ noise,
                 __bf16* __restrict__ xb, float* __restrict__ gates)
{
    const int b = blockIdx.x;
    const int t = threadIdx.x;            // 128 threads

    float cg[NEXP], cn[NEXP];
#pragma unroll
    for (int e = 0; e < NEXP; ++e) { cg[e] = 0.f; cn[e] = 0.f; }

    for (int d = t; d < DIN; d += 128) {
        float xv;
        if      (d < D_SF)                 xv = sf [(long long)b*D_SF  + d];
        else if (d < D_SF+D_OF)            xv = of [(long long)b*D_OF  + (d - D_SF)];
        else if (d < D_SF+D_OF+D_PPF)      xv = ppf[(long long)b*D_PPF + (d - D_SF - D_OF)];
        else                               xv = pvf[(long long)b*D_PVF + (d - D_SF - D_OF - D_PPF)];
        xb[(long long)b*DIN + d] = (__bf16)xv;

        const float* wg = w_gate  + (long long)d*NEXP;
        const float* wn = w_noise + (long long)d*NEXP;
#pragma unroll
        for (int e = 0; e < NEXP; ++e) { cg[e] += xv * wg[e]; cn[e] += xv * wn[e]; }
    }

    __shared__ float sg[NEXP][128];
    __shared__ float sn[NEXP][128];
#pragma unroll
    for (int e = 0; e < NEXP; ++e) { sg[e][t] = cg[e]; sn[e][t] = cn[e]; }
    __syncthreads();

    if (t < NEXP) {
        float a = 0.f, s = 0.f;
        for (int i = 0; i < 128; ++i) { a += sg[t][i]; s += sn[t][i]; }
        sg[t][0] = a; sn[t][0] = s;
    }
    __syncthreads();

    if (t == 0) {
        float logit[NEXP];
#pragma unroll
        for (int e = 0; e < NEXP; ++e) {
            float z  = sn[e][0];
            float sp = (z > 0.f) ? (z + log1pf(expf(-z))) : log1pf(expf(z));  // softplus
            logit[e] = sg[e][0] + noise[(long long)b*NEXP + e] * (sp + 1e-2f);
        }
        int   idx[KTOP];
        float val[KTOP];
        bool  used[NEXP];
#pragma unroll
        for (int e = 0; e < NEXP; ++e) used[e] = false;
        for (int k = 0; k < KTOP; ++k) {
            int   bi = -1; float bv = -3.4e38f;
            for (int e = 0; e < NEXP; ++e)
                if (!used[e] && logit[e] > bv) { bv = logit[e]; bi = e; }
            used[bi] = true; idx[k] = bi; val[k] = bv;
        }
        float mx = val[0], s = 0.f, ev[KTOP];
        for (int k = 0; k < KTOP; ++k) { ev[k] = expf(val[k] - mx); s += ev[k]; }
        float* g = gates + (long long)b*NEXP;
#pragma unroll
        for (int e = 0; e < NEXP; ++e) g[e] = 0.f;
        for (int k = 0; k < KTOP; ++k) g[idx[k]] = ev[k] / s;
    }
}

// =====================================================================
// Kernel 2: tiled fp32 -> bf16 transpose  (src [e][R][C] -> dst [e][C][R])
// =====================================================================
__global__ __launch_bounds__(256)
void transpose_f32_to_bf16(const float* __restrict__ src, __bf16* __restrict__ dst,
                           int R, int C, long long srcStride, long long dstStride)
{
    __shared__ float tile[32][33];
    const int e  = blockIdx.z;
    const float* S = src + (long long)e * srcStride;
    __bf16*      Dp = dst + (long long)e * dstStride;
    const int c0 = blockIdx.x * 32, r0 = blockIdx.y * 32;
    const int tx = threadIdx.x, ty = threadIdx.y;   // (32, 8)
#pragma unroll
    for (int j = 0; j < 32; j += 8)
        tile[ty + j][tx] = S[(long long)(r0 + ty + j) * C + (c0 + tx)];
    __syncthreads();
#pragma unroll
    for (int j = 0; j < 32; j += 8)
        Dp[(long long)(c0 + ty + j) * R + (r0 + tx)] = (__bf16)tile[tx][ty + j];
}

// =====================================================================
// Kernel 3: pack head weights / bias into padded layouts
// =====================================================================
__global__ __launch_bounds__(256)
void pack_heads(const float* __restrict__ Ws, const float* __restrict__ Wo,
                const float* __restrict__ Wp, __bf16* __restrict__ wht)
{
    long long i = (long long)blockIdx.x * blockDim.x + threadIdx.x;
    const long long tot = (long long)NEXP * C_PAD * HID;
    if (i >= tot) return;
    int h = (int)(i % HID);
    long long r = i / HID;
    int c = (int)(r % C_PAD);
    int e = (int)(r / C_PAD);
    float v = 0.f;
    if      (c < N_OBJ)         v = Ws[((long long)e*HID + h)*N_OBJ  + c];
    else if (c < 2*N_OBJ)       v = Wo[((long long)e*HID + h)*N_OBJ  + (c - N_OBJ)];
    else if (c < C_TOT)         v = Wp[((long long)e*HID + h)*N_PRED + (c - 2*N_OBJ)];
    wht[i] = (__bf16)v;
}

__global__ __launch_bounds__(256)
void pack_head_bias(const float* __restrict__ bs, const float* __restrict__ bo,
                    const float* __restrict__ bp, float* __restrict__ bh)
{
    int i = blockIdx.x * blockDim.x + threadIdx.x;
    if (i >= NEXP * C_PAD) return;
    int c = i % C_PAD, e = i / C_PAD;
    float v = 0.f;
    if      (c < N_OBJ)   v = bs[e*N_OBJ  + c];
    else if (c < 2*N_OBJ) v = bo[e*N_OBJ  + (c - N_OBJ)];
    else if (c < C_TOT)   v = bp[e*N_PRED + (c - 2*N_OBJ)];
    bh[i] = v;
}

// =====================================================================
// Kernel 4: LDS-staged, double-buffered bf16 WMMA GEMM
//   Out[e][M,N] = act(A[e][M,K] * Bt[e][N,K]^T + bias[e][N])
// Block: 256 thr = 8 waves (4 wm x 2 wn); wave tile 64x64; block tile 256x128.
// K staged in chunks of 64 through LDS (async global->LDS when available).
// Per k-step per wave: 16 ds_load_b128 : 16 v_wmma (1:1).
// =====================================================================
#define A_BYTES     36864                 // 256 rows x 72 bf16 (144B rows)
#define B_BYTES     18432                 // 128 rows x 72 bf16
#define STG_BYTES   (A_BYTES + B_BYTES)   // 55296 per stage, x2 = 108 KB LDS
#define A_OFF       0
#define B_OFF       A_BYTES
#define LDS_STRIDE  72                    // row stride in bf16 elements (conflict-free)

__device__ __forceinline__ bf16x16 ld_frag(const __bf16* p)
{
    bf16x8 lo = *(const bf16x8*)(p);
    bf16x8 hi = *(const bf16x8*)(p + 16);
    bf16x16 r;
#pragma unroll
    for (int i = 0; i < 8; ++i) { r[i] = lo[i]; r[i + 8] = hi[i]; }
    return r;
}

__global__ __launch_bounds__(256)
void gemm_wmma_bf16(const __bf16* __restrict__ A, long long aStrideE,
                    const __bf16* __restrict__ Bt, long long bStrideE,
                    const float*  __restrict__ bias, long long biasStrideE,
                    void* __restrict__ Out, long long oStrideE,
                    int M, int N, int Kd, int relu, int outBf16)
{
    __shared__ alignas(16) char smem[2 * STG_BYTES];

    const int e    = blockIdx.z;
    const __bf16* Ae = A  + (long long)e * aStrideE;
    const __bf16* Be = Bt + (long long)e * bStrideE;
    const float*  be = bias + (long long)e * biasStrideE;

    const int tid   = threadIdx.x;
    const int lane  = tid & 31;
    const int wid   = tid >> 5;          // 0..7
    const int wm    = wid >> 1;          // 0..3  -> M
    const int wn    = wid & 1;           // 0..1  -> N
    const int m0b   = blockIdx.y * 256;  // block M origin
    const int n0b   = blockIdx.x * 128;  // block N origin
    const int m0w   = wm * 64;           // wave M origin within block
    const int n0w   = wn * 64;           // wave N origin within block
    const int lrow  = lane & 15;
    const int lhalf = lane >> 4;
    const int koff  = lhalf * 8;         // lanes 16-31 take the +8 K half

    f32x8 acc[4][4];
#pragma unroll
    for (int i = 0; i < 4; ++i)
#pragma unroll
        for (int j = 0; j < 4; ++j)
#pragma unroll
            for (int v = 0; v < 8; ++v) acc[i][j][v] = 0.f;

    const int S = Kd >> 6;               // stages of K=64

    // ---- stage issue: 256x64 A tile + 128x64 B tile, 12 x 16B chunks / thread
    auto issue_stage = [&](int s, int buf) {
        char* base = smem + buf * STG_BYTES;
        const int ks = s << 6;
#pragma unroll
        for (int i = 0; i < 8; ++i) {            // A: 2048 chunks
            int c   = tid + i * 256;
            int row = c >> 3;
            int col = c & 7;
            const __bf16* ga = Ae + (long long)(m0b + row) * Kd + ks + col * 8;
            char*         la = base + A_OFF + row * (LDS_STRIDE * 2) + col * 16;
            CP16(ga, la);
        }
#pragma unroll
        for (int i = 0; i < 4; ++i) {            // B: 1024 chunks
            int c   = tid + i * 256;
            int row = c >> 3;
            int col = c & 7;
            const __bf16* gb = Be + (long long)(n0b + row) * Kd + ks + col * 8;
            char*         lb = base + B_OFF + row * (LDS_STRIDE * 2) + col * 16;
            CP16(gb, lb);
        }
    };

    issue_stage(0, 0);

    for (int s = 0; s < S; ++s) {
        const int buf = s & 1;
        if (s + 1 < S) {
            issue_stage(s + 1, buf ^ 1);
            S_WAIT_ASYNC(12);            // oldest 12 (= stage s) complete
        } else {
            S_WAIT_ASYNC(0);
        }
        __syncthreads();                 // stage s visible to all waves

        const __bf16* Ab = (const __bf16*)(smem + buf * STG_BYTES + A_OFF);
        const __bf16* Bb = (const __bf16*)(smem + buf * STG_BYTES + B_OFF);

#pragma unroll
        for (int kk = 0; kk < 64; kk += 32) {
            bf16x16 af[4], bfr[4];
#pragma unroll
            for (int mt = 0; mt < 4; ++mt)
                af[mt] = ld_frag(Ab + (m0w + mt*16 + lrow) * LDS_STRIDE + kk + koff);
#pragma unroll
            for (int nt = 0; nt < 4; ++nt)
                bfr[nt] = ld_frag(Bb + (n0w + nt*16 + lrow) * LDS_STRIDE + kk + koff);
#pragma unroll
            for (int mt = 0; mt < 4; ++mt)
#pragma unroll
                for (int nt = 0; nt < 4; ++nt)
                    acc[mt][nt] = __builtin_amdgcn_wmma_f32_16x16x32_bf16(
                        false, af[mt], false, bfr[nt], (short)0, acc[mt][nt], false, false);
        }
        __syncthreads();                 // all waves done with buf before reuse
    }

    // ---- epilogue: bias + act; two 32-row passes through LDS, coalesced stores
    float bv[4];
#pragma unroll
    for (int nt = 0; nt < 4; ++nt) bv[nt] = be[n0b + n0w + nt*16 + lrow];

#pragma unroll
    for (int half = 0; half < 2; ++half) {
        if (outBf16) {
            __bf16* tile = (__bf16*)(smem + wid * 4608);   // 32 x 72 bf16 region
#pragma unroll
            for (int mt2 = 0; mt2 < 2; ++mt2)
#pragma unroll
                for (int nt = 0; nt < 4; ++nt)
#pragma unroll
                    for (int v = 0; v < 8; ++v) {
                        float x = acc[half*2 + mt2][nt][v] + bv[nt];
                        if (relu) x = fmaxf(x, 0.f);
                        tile[(mt2*16 + v + lhalf*8) * LDS_STRIDE + nt*16 + lrow] = (__bf16)x;
                    }
            __syncthreads();
#pragma unroll
            for (int i = 0; i < 8; ++i) {
                int row   = i * 4 + (lane >> 3);   // 0..31
                int chunk = lane & 7;              // 0..7  (8 bf16 each)
                __bf16* op = (__bf16*)Out + (long long)e * oStrideE
                           + (long long)(m0b + m0w + half*32 + row) * N + n0b + n0w + chunk * 8;
                *(bf16x8*)op = *(const bf16x8*)(tile + row * LDS_STRIDE + chunk * 8);
            }
            __syncthreads();
        } else {
            float* tile = (float*)(smem + wid * 8704);     // 32 x 68 f32 region
#pragma unroll
            for (int mt2 = 0; mt2 < 2; ++mt2)
#pragma unroll
                for (int nt = 0; nt < 4; ++nt)
#pragma unroll
                    for (int v = 0; v < 8; ++v) {
                        float x = acc[half*2 + mt2][nt][v] + bv[nt];
                        if (relu) x = fmaxf(x, 0.f);
                        tile[(mt2*16 + v + lhalf*8) * 68 + nt*16 + lrow] = x;
                    }
            __syncthreads();
#pragma unroll
            for (int i = 0; i < 16; ++i) {
                int row   = i * 2 + (lane >> 4);   // 0..31
                int chunk = lane & 15;             // 0..15 (4 f32 each)
                float* op = (float*)Out + (long long)e * oStrideE
                          + (long long)(m0b + m0w + half*32 + row) * N + n0b + n0w + chunk * 4;
                *(f32x4*)op = *(const f32x4*)(tile + row * 68 + chunk * 4);
            }
            __syncthreads();
        }
    }
}

// =====================================================================
// Kernel 5: per-token softmax-per-group, gate-weighted combine, log
// =====================================================================
__global__ __launch_bounds__(256)
void combine_kernel(const float* __restrict__ hl, const float* __restrict__ gates,
                    float* __restrict__ out)
{
    const int b = blockIdx.x;
    const int t = threadIdx.x;            // 256
    __shared__ float red[C_TOT];
    __shared__ float logZ[3];
    __shared__ float gsh[NEXP];
    if (t < NEXP) gsh[t] = gates[(long long)b*NEXP + t];

    float comb = 0.f;
    for (int e = 0; e < NEXP; ++e) {
        __syncthreads();
        if (t < C_TOT) red[t] = hl[((long long)e*BTOK + b)*C_PAD + t];
        __syncthreads();
        if (t < 3) {
            int s0 = (t == 0) ? 0 : (t == 1) ? N_OBJ : 2*N_OBJ;
            int s1 = (t == 0) ? N_OBJ : (t == 1) ? 2*N_OBJ : C_TOT;
            float m = -3.4e38f;
            for (int i = s0; i < s1; ++i) m = fmaxf(m, red[i]);
            float s = 0.f;
            for (int i = s0; i < s1; ++i) s += expf(red[i] - m);
            logZ[t] = m + logf(s);
        }
        __syncthreads();
        if (t < C_TOT) {
            int g = (t < N_OBJ) ? 0 : (t < 2*N_OBJ) ? 1 : 2;
            comb += gsh[e] * expf(red[t] - logZ[g]);
        }
    }
    if (t < C_TOT) {
        float r = logf(comb == 0.f ? EPS64 : comb);
        if      (t < N_OBJ)   out[(long long)b*N_OBJ + t] = r;
        else if (t < 2*N_OBJ) out[(long long)BTOK*N_OBJ   + (long long)b*N_OBJ  + (t - N_OBJ)]   = r;
        else                  out[(long long)BTOK*2*N_OBJ + (long long)b*N_PRED + (t - 2*N_OBJ)] = r;
    }
}

// =====================================================================
// Host launcher
// =====================================================================
extern "C" void kernel_launch(void* const* d_in, const int* in_sizes, int n_in,
                              void* d_out, int out_size, void* d_ws, size_t ws_size,
                              hipStream_t stream)
{
    const float* sf      = (const float*)d_in[0];
    const float* of      = (const float*)d_in[1];
    const float* ppf     = (const float*)d_in[2];
    const float* pvf     = (const float*)d_in[3];
    const float* w_gate  = (const float*)d_in[4];
    const float* w_noise = (const float*)d_in[5];
    const float* W1      = (const float*)d_in[6];
    const float* b1      = (const float*)d_in[7];
    const float* W2      = (const float*)d_in[8];
    const float* b2      = (const float*)d_in[9];
    const float* Ws      = (const float*)d_in[10];
    const float* bs      = (const float*)d_in[11];
    const float* Wo      = (const float*)d_in[12];
    const float* bo      = (const float*)d_in[13];
    const float* Wp      = (const float*)d_in[14];
    const float* bp      = (const float*)d_in[15];
    const float* noise   = (const float*)d_in[16];

    char* ws = (char*)d_ws;
    __bf16* xb    = (__bf16*)(ws + OFF_XB);
    float*  gates = (float*) (ws + OFF_GATES);
    __bf16* w1t   = (__bf16*)(ws + OFF_W1T);
    __bf16* w2t   = (__bf16*)(ws + OFF_W2T);
    __bf16* wht   = (__bf16*)(ws + OFF_WHT);
    float*  bh    = (float*) (ws + OFF_BH);
    __bf16* h1    = (__bf16*)(ws + OFF_H1);
    __bf16* h2    = (__bf16*)(ws + OFF_H2);
    float*  hl    = (float*) (ws + OFF_HL);
    float*  outp  = (float*)d_out;

    // 1) weight prep
    transpose_f32_to_bf16<<<dim3(HID/32, DIN/32, NEXP), dim3(32, 8), 0, stream>>>(
        W1, w1t, DIN, HID, (long long)DIN*HID, (long long)HID*DIN);
    transpose_f32_to_bf16<<<dim3(HID/32, HID/32, NEXP), dim3(32, 8), 0, stream>>>(
        W2, w2t, HID, HID, (long long)HID*HID, (long long)HID*HID);
    {
        long long tot = (long long)NEXP * C_PAD * HID;
        pack_heads<<<(unsigned)((tot + 255) / 256), 256, 0, stream>>>(Ws, Wo, Wp, wht);
        pack_head_bias<<<(NEXP*C_PAD + 255) / 256, 256, 0, stream>>>(bs, bo, bp, bh);
    }

    // 2) gating + x concat/convert
    gate_kernel<<<BTOK, 128, 0, stream>>>(sf, of, ppf, pvf, w_gate, w_noise, noise, xb, gates);

    // 3) h1 = relu(x @ W1 + b1)  -> bf16 [E,B,H]
    gemm_wmma_bf16<<<dim3(HID/128, BTOK/256, NEXP), 256, 0, stream>>>(
        xb, 0LL, w1t, (long long)HID*DIN, b1, (long long)HID,
        (void*)h1, (long long)BTOK*HID, BTOK, HID, DIN, /*relu=*/1, /*outBf16=*/1);

    // 4) h2 = relu(h1 @ W2 + b2) -> bf16 [E,B,H]
    gemm_wmma_bf16<<<dim3(HID/128, BTOK/256, NEXP), 256, 0, stream>>>(
        h1, (long long)BTOK*HID, w2t, (long long)HID*HID, b2, (long long)HID,
        (void*)h2, (long long)BTOK*HID, BTOK, HID, HID, /*relu=*/1, /*outBf16=*/1);

    // 5) head logits = h2 @ Wh + bh -> f32 [E,B,C_PAD]
    gemm_wmma_bf16<<<dim3(C_PAD/128, BTOK/256, NEXP), 256, 0, stream>>>(
        h2, (long long)BTOK*HID, wht, (long long)C_PAD*HID, bh, (long long)C_PAD,
        (void*)hl, (long long)BTOK*C_PAD, BTOK, C_PAD, HID, /*relu=*/0, /*outBf16=*/0);

    // 6) softmax-combine + log
    combine_kernel<<<BTOK, 256, 0, stream>>>(hl, gates, outp);
}
